// RNN_75797582840480
// MI455X (gfx1250) — matrix operand order repaired and checked by reading, not compiled, not run on hardware
//
#include <hip/hip_runtime.h>

#define B_ 64
#define T_ 2048
#define I_ 256
#define H_ 256
#define G_ 1024
#define HSTR 272   // padded LDS row stride (bf16 elems): 544B rows, 16B aligned

typedef __attribute__((ext_vector_type(16))) __bf16 v16bf;
typedef __attribute__((ext_vector_type(8)))  __bf16 v8bf;
typedef __attribute__((ext_vector_type(4)))  __bf16 v4bf;
typedef __attribute__((ext_vector_type(8)))  float  v8f;
typedef __attribute__((ext_vector_type(4)))  float  v4f;

// Wx/Wh [K=256][N=1024] f32 -> WxT/WhT [N=1024][K=256] bf16 (B-fragments contiguous).
__global__ void prep_weights(const float* __restrict__ Wx,
                             const float* __restrict__ Wh,
                             __bf16* __restrict__ WxT,
                             __bf16* __restrict__ WhT) {
  int idx = blockIdx.x * blockDim.x + threadIdx.x;  // 0 .. 1024*256-1
  int n = idx >> 8;
  int k = idx & 255;
  WxT[idx] = (__bf16)Wx[k * G_ + n];
  WhT[idx] = (__bf16)Wh[k * G_ + n];
}

// x f32 -> bf16 (one-shot, bandwidth-trivial).
__global__ void prep_x(const float* __restrict__ x, __bf16* __restrict__ xb) {
  size_t i = ((size_t)blockIdx.x * blockDim.x + threadIdx.x) * 4;
  v4f f = *(const v4f*)(x + i);
  v4bf o;
#pragma unroll
  for (int j = 0; j < 4; ++j) o[j] = (__bf16)f[j];
  *(v4bf*)(xb + i) = o;
}

__device__ __forceinline__ float sigf(float x)     { return 1.f / (1.f + __expf(-x)); }
__device__ __forceinline__ float tanhfast(float x) { return 1.f - 2.f / (__expf(2.f * x) + 1.f); }

// 4 workgroups (16 batch rows each) x 16 waves. Wave w owns columns w*16..w*16+15
// of all four gates; c/h tile stays in registers; h exchanged via double-buffered LDS.
// `wstep` is 0 at runtime; defeats LICM so weight fragments stream from L2 each step.
template <bool XB>
__global__ __launch_bounds__(512) void lstm_scan(
    const float* __restrict__ x, const __bf16* __restrict__ xb,
    const __bf16* __restrict__ WxT, const __bf16* __restrict__ WhT,
    const float* __restrict__ bias_g,
    const float* __restrict__ h0, const float* __restrict__ c0,
    float* __restrict__ out, long wstep) {
  const int tid  = threadIdx.x;
  const int wave = tid >> 5;
  const int lane = tid & 31;
  const int lm   = lane & 15;   // A row m / B col n / D col n
  const int hi   = lane >> 4;   // lane-half selector in WMMA layouts
  const int b0   = blockIdx.x << 4;
  const int ncol = (wave << 4) + lm;

  __shared__ __bf16 hbuf[2][16 * HSTR];

  float bias[4];
#pragma unroll
  for (int g = 0; g < 4; ++g) bias[g] = bias_g[g * H_ + ncol];

  // c in registers, WMMA D layout: comp r <-> row m = r + 8*hi, col = ncol.
  v8f c;
#pragma unroll
  for (int r = 0; r < 8; ++r)
    c[r] = c0[(size_t)(b0 + r + 8 * hi) * H_ + ncol];

  for (int idx = tid; idx < 16 * H_; idx += 512) {
    int m = idx >> 8, n = idx & 255;
    hbuf[0][m * HSTR + n] = (__bf16)h0[(size_t)(b0 + m) * H_ + n];
  }
  __syncthreads();

  // Per-wave weight base pointers (advanced by runtime-zero wstep each step).
  const __bf16* wx = WxT + (size_t)ncol * I_ + hi * 8;
  const __bf16* wh = WhT + (size_t)ncol * H_ + hi * 8;

  const float*  xrow  = x  + (size_t)(b0 + lm) * T_ * I_;
  const __bf16* xbrow = XB ? (xb + (size_t)(b0 + lm) * T_ * I_) : nullptr;

  for (int t = 0; t < T_; ++t) {
    const int cur = t & 1;

    v8f acc[4];
#pragma unroll
    for (int g = 0; g < 4; ++g)
#pragma unroll
      for (int r = 0; r < 8; ++r) acc[g][r] = 0.f;   // zero-seed: inline-const C

    // ---- x_t @ Wx : K=256 in 8 steps of 32 ----
#pragma unroll
    for (int ks = 0; ks < 8; ++ks) {
      v16bf a;
      if (XB) {
        const __bf16* ap = xbrow + (size_t)t * I_ + ks * 32 + hi * 8;
        if (ks == 0 && t + 1 < T_) __builtin_prefetch(ap + I_, 0, 0);
        v8bf a0 = *(const v8bf*)(ap);
        v8bf a1 = *(const v8bf*)(ap + 16);
#pragma unroll
        for (int j = 0; j < 8; ++j) { a[j] = a0[j]; a[j + 8] = a1[j]; }
      } else {
        const float* ap = xrow + (size_t)t * I_ + ks * 32 + hi * 8;
        if (ks == 0 && t + 1 < T_) __builtin_prefetch(ap + I_, 0, 0);
        v4f f0 = *(const v4f*)(ap);
        v4f f1 = *(const v4f*)(ap + 4);
        v4f f2 = *(const v4f*)(ap + 16);
        v4f f3 = *(const v4f*)(ap + 20);
#pragma unroll
        for (int j = 0; j < 4; ++j) {
          a[j] = (__bf16)f0[j];  a[j + 4]  = (__bf16)f1[j];
          a[j + 8] = (__bf16)f2[j];  a[j + 12] = (__bf16)f3[j];
        }
      }
      // Batch all 4 gate B-fragments first -> 8-load clause, staged waits,
      // so 3 of 4 load latencies overlap with WMMA issue.
      v16bf bm[4];
#pragma unroll
      for (int g = 0; g < 4; ++g) {
        const __bf16* wp = wx + (size_t)g * (H_ * I_) + ks * 32;
        v8bf blo = *(const v8bf*)wp;
        v8bf bhi = *(const v8bf*)(wp + 16);
#pragma unroll
        for (int j = 0; j < 8; ++j) { bm[g][j] = blo[j]; bm[g][j + 8] = bhi[j]; }
      }
#pragma unroll
      for (int g = 0; g < 4; ++g)
        acc[g] = __builtin_amdgcn_wmma_f32_16x16x32_bf16(
            false, a, false, bm[g], (short)0, acc[g], false, false);
    }

    // ---- h_t @ Wh : h from double-buffered LDS (bf16) ----
    const __bf16* hrow = &hbuf[cur][lm * HSTR];
#pragma unroll
    for (int ks = 0; ks < 8; ++ks) {
      const __bf16* ap = hrow + ks * 32 + hi * 8;
      v8bf a0 = *(const v8bf*)(ap);
      v8bf a1 = *(const v8bf*)(ap + 16);
      v16bf a;
#pragma unroll
      for (int j = 0; j < 8; ++j) { a[j] = a0[j]; a[j + 8] = a1[j]; }
      v16bf bm[4];
#pragma unroll
      for (int g = 0; g < 4; ++g) {
        const __bf16* wp = wh + (size_t)g * (H_ * H_) + ks * 32;
        v8bf blo = *(const v8bf*)wp;
        v8bf bhi = *(const v8bf*)(wp + 16);
#pragma unroll
        for (int j = 0; j < 8; ++j) { bm[g][j] = blo[j]; bm[g][j + 8] = bhi[j]; }
      }
#pragma unroll
      for (int g = 0; g < 4; ++g)
        acc[g] = __builtin_amdgcn_wmma_f32_16x16x32_bf16(
            false, a, false, bm[g], (short)0, acc[g], false, false);
    }

    // ---- gates + state update; write h to LDS + global out ----
#pragma unroll
    for (int r = 0; r < 8; ++r) {
      float ig = sigf(acc[0][r] + bias[0]);
      float fg = sigf(acc[1][r] + bias[1]);
      float gg = tanhfast(acc[2][r] + bias[2]);
      float og = sigf(acc[3][r] + bias[3]);
      float cn = fg * c[r] + ig * gg;
      c[r] = cn;
      float hn = og * tanhfast(cn);
      int m = r + 8 * hi;
      hbuf[cur ^ 1][m * HSTR + ncol] = (__bf16)hn;
      out[((size_t)(b0 + m) * T_ + t) * H_ + ncol] = hn;
    }

    // Runtime-zero stride: keeps weight loads non-invariant (no LICM hoist/spill).
    wx += wstep;
    wh += wstep;
    __syncthreads();  // one barrier per timestep (double-buffered h)
  }
}

extern "C" void kernel_launch(void* const* d_in, const int* in_sizes, int n_in,
                              void* d_out, int out_size, void* d_ws, size_t ws_size,
                              hipStream_t stream) {
  const float* x  = (const float*)d_in[0];
  const float* Wx = (const float*)d_in[1];
  const float* Wh = (const float*)d_in[2];
  const float* bg = (const float*)d_in[3];
  const float* h0 = (const float*)d_in[4];
  const float* c0 = (const float*)d_in[5];
  float* out = (float*)d_out;

  const size_t WMAT   = (size_t)G_ * I_ * sizeof(__bf16);        // 512 KB each
  const size_t XBYTES = (size_t)B_ * T_ * I_ * sizeof(__bf16);   // 64 MB

  __bf16* WxT = (__bf16*)d_ws;
  __bf16* WhT = (__bf16*)((char*)d_ws + WMAT);
  __bf16* xbf = (__bf16*)((char*)d_ws + 2 * WMAT);

  prep_weights<<<(G_ * I_) / 256, 256, 0, stream>>>(Wx, Wh, WxT, WhT);

  if (ws_size >= 2 * WMAT + XBYTES) {
    prep_x<<<(B_ * T_ * I_) / (256 * 4), 256, 0, stream>>>(x, xbf);
    lstm_scan<true><<<B_ / 16, 512, 0, stream>>>(x, xbf, WxT, WhT, bg, h0, c0,
                                                 out, 0L);
  } else {
    lstm_scan<false><<<B_ / 16, 512, 0, stream>>>(x, nullptr, WxT, WhT, bg, h0,
                                                  c0, out, 0L);
  }
}
